// SwinTransformerLayer_58377195487711
// MI455X (gfx1250) — matrix operand order repaired and checked by reading, not compiled, not run on hardware
//
#include <hip/hip_runtime.h>
#include <hip/hip_bf16.h>
#include <math.h>

// ---------------------------------------------------------------------------
// Swin Transformer layer (shifted-window MHA + MLP) for MI455X / gfx1250.
// All matmuls run on v_wmma_f32_16x16x32_bf16 (wave32 WMMA, f32 accumulate).
// ---------------------------------------------------------------------------

typedef __bf16 bf16;
typedef __attribute__((ext_vector_type(16))) __bf16 v16bf;
typedef __attribute__((ext_vector_type(8)))  __bf16 v8bf;
typedef __attribute__((ext_vector_type(4)))  __bf16 v4bf;
typedef __attribute__((ext_vector_type(8)))  float  v8f;

#define DEV __device__ __forceinline__

constexpr int   Bn    = 8;
constexpr int   Hh    = 112;
constexpr int   Wd    = 112;
constexpr int   Cc    = 128;
constexpr int   WS    = 7;
constexpr int   HEADS = 4;
constexpr int   HD    = 32;          // head dim (== WMMA K for QK^T!)
constexpr int   NTK   = 49;          // tokens / window
constexpr int   NWIN  = 256;         // windows / image
constexpr int   NTOK  = Bn * Hh * Wd;     // 100352 rows (multiple of 16)
constexpr int   NWTOT = Bn * NWIN;        // 2048 windows
constexpr float SCALE = 0.17677669529663689f;  // 32^-0.5
constexpr float LNEPS = 1e-3f;

// ---------------- WMMA helpers (CDNA5 wave32 fragment layouts) -------------

DEV v8f wmma_bf16(v16bf a, v16bf b, v8f c) {
  return __builtin_amdgcn_wmma_f32_16x16x32_bf16(false, a, false, b,
                                                 (short)0, c, false, false);
}

// A (16x32, row-major, lda = K): lanes 0-15 row M=lane, K {k..k+7, k+16..k+23};
// lanes 16-31 same row, K {k+8..k+15, k+24..k+31}.  Two contiguous 16B loads.
DEV v16bf load_a_frag(const bf16* A, int lda, int row0, int k0) {
  const int lane = threadIdx.x & 31;
  const bf16* p = A + (size_t)(row0 + (lane & 15)) * lda + k0 + ((lane & 16) ? 8 : 0);
  v8bf lo = *(const v8bf*)p;
  v8bf hi = *(const v8bf*)(p + 16);
  return __builtin_shufflevector(lo, hi, 0,1,2,3,4,5,6,7,8,9,10,11,12,13,14,15);
}

// B (32x16) taken from B^T (row-major [N,K], ldb = K): lane n<16 reads
// B^T[n0+n] K {k0..k0+7, k0+16..k0+23}; lanes 16-31 K {k0+8..k0+15, k0+24..k0+31}.
DEV v16bf load_b_frag(const bf16* Bt, int ldb, int n0, int k0) {
  const int lane = threadIdx.x & 31;
  const bf16* p = Bt + (size_t)(n0 + (lane & 15)) * ldb + k0 + ((lane & 16) ? 8 : 0);
  v8bf lo = *(const v8bf*)p;
  v8bf hi = *(const v8bf*)(p + 16);
  return __builtin_shufflevector(lo, hi, 0,1,2,3,4,5,6,7,8,9,10,11,12,13,14,15);
}

// C/D 16x16 f32: element r -> row = r + (lane&16 ? 8 : 0), col = lane&15.

// One wave computes a 16 x (8*16) output tile; A fragment reused 8x per kstep.
template <int KD>
DEV void gemm16x128(const bf16* A, const bf16* Bt, int m0, int n0, v8f acc[8]) {
  v8f z = {};
#pragma unroll
  for (int nt = 0; nt < 8; nt++) acc[nt] = z;
  for (int k = 0; k < KD; k += 32) {
    v16bf a = load_a_frag(A, KD, m0, k);
    if (k + 32 < KD)  // hint next A chunk -> global_prefetch_b8
      __builtin_prefetch(A + (size_t)(m0 + (threadIdx.x & 15)) * KD + k + 32, 0, 0);
#pragma unroll
    for (int nt = 0; nt < 8; nt++) {
      v16bf b = load_b_frag(Bt, KD, n0 + nt * 16, k);
      acc[nt] = wmma_bf16(a, b, acc[nt]);
    }
  }
}

// ------------------------- index helpers -----------------------------------

DEV int region_of(int wr, int wc, int t) {   // shifted-image region id
  int i = t / 7, j = t % 7;
  int g = wr * 7 + i, q = wc * 7 + j;
  int rh = (g < 105) ? 0 : ((g < 109) ? 1 : 2);
  int rc = (q < 105) ? 0 : ((q < 109) ? 1 : 2);
  return rh * 3 + rc;
}

// token index (window-partition order) -> source pixel of the shifted image
DEV size_t token_to_pixel(int token) {
  int w = token / NTK, t = token % NTK;
  int b = w >> 8, win = w & 255, wr = win >> 4, wc = win & 15;
  int i = t / 7, j = t % 7;
  int hh = wr * 7 + i + 3; if (hh >= 112) hh -= 112;   // roll(-3) read side
  int ww = wc * 7 + j + 3; if (ww >= 112) ww -= 112;
  return (((size_t)b * 112 + hh) * 112 + ww) * (size_t)Cc;
}

// ---------------------- K0: weight convert / transpose ---------------------
// w_qkv[128,384] -> wqkvT[384,128]; w_proj[128,128] -> wpT[128,128];
// w1[128,512] -> w1T[512,128]; w2[512,128] -> w2T[128,512]   (all bf16)
__global__ void __launch_bounds__(256)
k_convert(const float* wqkv, const float* wproj, const float* w1, const float* w2,
          bf16* wqkvT, bf16* wpT, bf16* w1T, bf16* w2T) {
  int idx = blockIdx.x * 256 + threadIdx.x;
  if (idx < 49152) {                       // 384*128
    int n = idx / 128, k = idx % 128;
    wqkvT[idx] = (bf16)wqkv[k * 384 + n];
  } else if (idx < 49152 + 16384) {        // 128*128
    int o = idx - 49152; int n = o / 128, k = o % 128;
    wpT[o] = (bf16)wproj[k * 128 + n];
  } else if (idx < 49152 + 16384 + 65536) {// 512*128
    int o = idx - 49152 - 16384; int n = o / 128, k = o % 128;
    w1T[o] = (bf16)w1[k * 512 + n];
  } else if (idx < 196608) {               // 128*512
    int o = idx - 49152 - 16384 - 65536; int n = o / 512, k = o % 512;
    w2T[o] = (bf16)w2[k * 128 + n];
  }
}

// -------------------- LayerNorm core (1 wave per token) --------------------
DEV void ln_store(const float* row, const float* g, const float* be, bf16* dst) {
  const int lane = threadIdx.x & 31;
  const float4 v = *(const float4*)(row + lane * 4);
  float sm = v.x + v.y + v.z + v.w;
#pragma unroll
  for (int m = 1; m < 32; m <<= 1) sm += __shfl_xor(sm, m, 32);
  float mu = sm * 0.0078125f;
  float dx = v.x - mu, dy = v.y - mu, dz = v.z - mu, dw = v.w - mu;
  float sq = dx * dx + dy * dy + dz * dz + dw * dw;
#pragma unroll
  for (int m = 1; m < 32; m <<= 1) sq += __shfl_xor(sq, m, 32);
  float inv = rsqrtf(sq * 0.0078125f + LNEPS);
  const float4 gv = *(const float4*)(g + lane * 4);
  const float4 bv = *(const float4*)(be + lane * 4);
  v4bf o;
  o[0] = (bf16)(dx * inv * gv.x + bv.x);
  o[1] = (bf16)(dy * inv * gv.y + bv.y);
  o[2] = (bf16)(dz * inv * gv.z + bv.z);
  o[3] = (bf16)(dw * inv * gv.w + bv.w);
  *(v4bf*)(dst + lane * 4) = o;
}

// K1: LN1 + cyclic shift + window partition -> xw bf16 [NTOK,128]
__global__ void __launch_bounds__(256)
k_ln1(const float* x, const float* g1, const float* b1, bf16* xw) {
  int token = blockIdx.x * 8 + (threadIdx.x >> 5);
  ln_store(x + token_to_pixel(token), g1, b1, xw + (size_t)token * Cc);
}

// K5: LN2 over x1 (flat pixel rows) -> h2 bf16
__global__ void __launch_bounds__(256)
k_ln2(const float* x1, const float* g2, const float* b2, bf16* h2) {
  int token = blockIdx.x * 8 + (threadIdx.x >> 5);
  ln_store(x1 + (size_t)token * Cc, g2, b2, h2 + (size_t)token * Cc);
}

// -------------------- K2: QKV GEMM + scatter to per-head layout ------------
// out: q/k/v [window][head][49][32] bf16 ; q pre-scaled by SCALE
__global__ void __launch_bounds__(32)
k_qkv(const bf16* xw, const bf16* wqkvT, const float* bqkv,
      bf16* qb, bf16* kb, bf16* vb) {
  const int m0 = blockIdx.x * 16;
  const int part = blockIdx.y;             // 0=q 1=k 2=v
  const int n0 = part * 128;
  v8f acc[8];
  gemm16x128<128>(xw, wqkvT, m0, n0, acc);
  const int lane = threadIdx.x & 31;
  const int rsel = (lane & 16) ? 8 : 0;
  bf16* dst = (part == 0) ? qb : (part == 1) ? kb : vb;
  const float sc = (part == 0) ? SCALE : 1.0f;
#pragma unroll
  for (int nt = 0; nt < 8; nt++) {
    int cc = nt * 16 + (lane & 15);        // 0..127 within part
    int head = cc >> 5, d = cc & 31;
    float bias = bqkv[n0 + cc];
#pragma unroll
    for (int r = 0; r < 8; r++) {
      int m = m0 + r + rsel;
      int w = m / NTK, t = m % NTK;
      dst[(((size_t)w * HEADS + head) * NTK + t) * HD + d] =
          (bf16)((acc[nt][r] + bias) * sc);
    }
  }
}

// -------------------- K3: windowed attention (1 block = 1 window) ----------
// LDS: qs[4][64][32] | ks[4][64][32] | vts[4][32][64]; P reuses qs+ks region.
__global__ void __launch_bounds__(256)
k_attn(const bf16* qb, const bf16* kb, const bf16* vb,
       const float* bias_table, bf16* ob) {
  extern __shared__ bf16 smem[];
  bf16* qs  = smem;              // 8192 elems
  bf16* ks  = smem + 8192;       // 8192 elems
  bf16* vts = smem + 16384;      // 8192 elems
  bf16* ps  = smem;              // 16384 elems, reuses qs+ks after barrier

  const int w = blockIdx.x;
  const int win = w & 255, wr = win >> 4, wc = win & 15;
  const int tid = threadIdx.x;

  for (int idx = tid; idx < 8192; idx += 256) {       // q,k: [h][t(64)][d(32)]
    int head = idx >> 11, t = (idx >> 5) & 63, d = idx & 31;
    size_t g = (((size_t)w * HEADS + head) * NTK + t) * HD + d;
    bool ok = t < NTK;
    qs[idx] = ok ? qb[g] : (bf16)0.0f;
    ks[idx] = ok ? kb[g] : (bf16)0.0f;
  }
  for (int idx = tid; idx < 8192; idx += 256) {       // vT: [h][d(32)][t(64)]
    int head = idx >> 11, d = (idx >> 6) & 31, t = idx & 63;
    vts[idx] = (t < NTK)
        ? vb[(((size_t)w * HEADS + head) * NTK + t) * HD + d] : (bf16)0.0f;
  }
  __syncthreads();

  const int wave = tid >> 5, lane = tid & 31;
  const int head = wave >> 1, half = wave & 1;        // 2 waves per head
  const int colb = lane & 15, rsel = (lane & 16) ? 8 : 0;
  const bf16* qh = qs + head * 2048;
  const bf16* kh = ks + head * 2048;

  // S = q * k^T : head dim 32 == WMMA K -> single WMMA per 16x16 tile
  v8f s[2][4];
  v8f z = {};
#pragma unroll
  for (int mt2 = 0; mt2 < 2; mt2++) {
    int mt = half * 2 + mt2;
    v16bf a = load_a_frag(qh, 32, mt * 16, 0);
#pragma unroll
    for (int nt = 0; nt < 4; nt++) {
      v16bf b = load_b_frag(kh, 32, nt * 16, 0);      // B^T == k row-major
      s[mt2][nt] = wmma_bf16(a, b, z);
    }
  }

  // relative-position bias + shift mask (computed analytically, no tables)
#pragma unroll
  for (int mt2 = 0; mt2 < 2; mt2++)
#pragma unroll
    for (int nt = 0; nt < 4; nt++)
#pragma unroll
      for (int r = 0; r < 8; r++) {
        int row = (half * 2 + mt2) * 16 + r + rsel;
        int col = nt * 16 + colb;
        float sv = s[mt2][nt][r];
        if (col >= NTK) {
          sv = -1e30f;                                  // padded keys
        } else {
          int rn = row < NTK ? row : NTK - 1;           // padded queries: dummy
          int i1 = rn / 7, j1 = rn % 7, i2 = col / 7, j2 = col % 7;
          int bidx = (i1 - i2 + 6) * 13 + (j1 - j2 + 6);
          sv += bias_table[bidx * HEADS + head];
          if (region_of(wr, wc, rn) != region_of(wr, wc, col)) sv -= 100.0f;
        }
        s[mt2][nt][r] = sv;
      }

  __syncthreads();   // all waves done reading qs/ks before P overwrites them

  // row softmax in C-fragment registers; N lives in 16-lane groups
  bf16* ph = ps + head * 4096;
#pragma unroll
  for (int mt2 = 0; mt2 < 2; mt2++) {
    int mt = half * 2 + mt2;
#pragma unroll
    for (int r = 0; r < 8; r++) {
      float mx = fmaxf(fmaxf(s[mt2][0][r], s[mt2][1][r]),
                       fmaxf(s[mt2][2][r], s[mt2][3][r]));
#pragma unroll
      for (int m = 1; m < 16; m <<= 1) mx = fmaxf(mx, __shfl_xor(mx, m, 32));
      float p0 = __expf(s[mt2][0][r] - mx);
      float p1 = __expf(s[mt2][1][r] - mx);
      float p2 = __expf(s[mt2][2][r] - mx);
      float p3 = __expf(s[mt2][3][r] - mx);
      float sum = p0 + p1 + p2 + p3;
#pragma unroll
      for (int m = 1; m < 16; m <<= 1) sum += __shfl_xor(sum, m, 32);
      float inv = 1.0f / sum;
      int row = mt * 16 + r + rsel;
      ph[row * 64 + 0 * 16 + colb] = (bf16)(p0 * inv);
      ph[row * 64 + 1 * 16 + colb] = (bf16)(p1 * inv);
      ph[row * 64 + 2 * 16 + colb] = (bf16)(p2 * inv);
      ph[row * 64 + 3 * 16 + colb] = (bf16)(p3 * inv);
    }
  }
  __syncthreads();

  // O = P * V : K = 64 (2 ksteps), N = 32 (2 col tiles)
  const bf16* vh = vts + head * 2048;                  // vT [32][64]
  v8f o[2][2];
#pragma unroll
  for (int i = 0; i < 2; i++)
#pragma unroll
    for (int j = 0; j < 2; j++) o[i][j] = z;
#pragma unroll
  for (int k0 = 0; k0 < 64; k0 += 32)
#pragma unroll
    for (int mt2 = 0; mt2 < 2; mt2++) {
      v16bf a = load_a_frag(ph, 64, (half * 2 + mt2) * 16, k0);
#pragma unroll
      for (int ct = 0; ct < 2; ct++) {
        v16bf b = load_b_frag(vh, 64, ct * 16, k0);    // B^T == vT row-major
        o[mt2][ct] = wmma_bf16(a, b, o[mt2][ct]);
      }
    }

#pragma unroll
  for (int mt2 = 0; mt2 < 2; mt2++)
#pragma unroll
    for (int ct = 0; ct < 2; ct++)
#pragma unroll
      for (int r = 0; r < 8; r++) {
        int trow = (half * 2 + mt2) * 16 + r + rsel;
        if (trow < NTK) {
          int c = head * HD + ct * 16 + colb;
          ob[((size_t)w * NTK + trow) * Cc + c] = (bf16)o[mt2][ct][r];
        }
      }
}

// ----- K4: proj GEMM + window reverse + roll(+3) + residual -> x1 f32 ------
__global__ void __launch_bounds__(32)
k_proj(const bf16* ob, const bf16* wpT, const float* bp,
       const float* x, float* x1) {
  const int m0 = blockIdx.x * 16;
  v8f acc[8];
  gemm16x128<128>(ob, wpT, m0, 0, acc);
  const int lane = threadIdx.x & 31;
  const int rsel = (lane & 16) ? 8 : 0;
#pragma unroll
  for (int r = 0; r < 8; r++) {
    size_t base = token_to_pixel(m0 + r + rsel);
#pragma unroll
    for (int nt = 0; nt < 8; nt++) {
      int c = nt * 16 + (lane & 15);
      x1[base + c] = x[base + c] + acc[nt][r] + bp[c];
    }
  }
}

// ----- K6: MLP1 GEMM + exact GELU -> m1 bf16 [NTOK,512] --------------------
__global__ void __launch_bounds__(32)
k_mlp1(const bf16* h2, const bf16* w1T, const float* b1, bf16* m1) {
  const int m0 = blockIdx.x * 16;
  const int n0 = blockIdx.y * 128;
  v8f acc[8];
  gemm16x128<128>(h2, w1T, m0, n0, acc);
  const int lane = threadIdx.x & 31;
  const int rsel = (lane & 16) ? 8 : 0;
#pragma unroll
  for (int nt = 0; nt < 8; nt++) {
    int c = n0 + nt * 16 + (lane & 15);
    float bias = b1[c];
#pragma unroll
    for (int r = 0; r < 8; r++) {
      float v = acc[nt][r] + bias;
      float g = 0.5f * v * (1.0f + erff(v * 0.70710678118654752f));
      m1[(size_t)(m0 + r + rsel) * 512 + c] = (bf16)g;
    }
  }
}

// ----- K7: MLP2 GEMM (K=512) + residual -> out f32 -------------------------
__global__ void __launch_bounds__(32)
k_mlp2(const bf16* m1, const bf16* w2T, const float* b2,
       const float* x1, float* out) {
  const int m0 = blockIdx.x * 16;
  v8f acc[8];
  gemm16x128<512>(m1, w2T, m0, 0, acc);
  const int lane = threadIdx.x & 31;
  const int rsel = (lane & 16) ? 8 : 0;
#pragma unroll
  for (int r = 0; r < 8; r++) {
    size_t rowb = (size_t)(m0 + r + rsel) * Cc;
#pragma unroll
    for (int nt = 0; nt < 8; nt++) {
      int c = nt * 16 + (lane & 15);
      out[rowb + c] = x1[rowb + c] + acc[nt][r] + b2[c];
    }
  }
}

// ---------------------------------------------------------------------------

extern "C" void kernel_launch(void* const* d_in, const int* in_sizes, int n_in,
                              void* d_out, int out_size, void* d_ws, size_t ws_size,
                              hipStream_t stream) {
  const float* x       = (const float*)d_in[0];
  const float* gamma1  = (const float*)d_in[1];
  const float* beta1   = (const float*)d_in[2];
  const float* w_qkv   = (const float*)d_in[3];
  const float* b_qkv   = (const float*)d_in[4];
  const float* btab    = (const float*)d_in[5];
  const float* w_proj  = (const float*)d_in[6];
  const float* b_proj  = (const float*)d_in[7];
  const float* gamma2  = (const float*)d_in[8];
  const float* beta2   = (const float*)d_in[9];
  const float* w1      = (const float*)d_in[10];
  const float* b1      = (const float*)d_in[11];
  const float* w2      = (const float*)d_in[12];
  const float* b2      = (const float*)d_in[13];
  float* out = (float*)d_out;

  // workspace layout (bytes, all 4KB aligned)
  char* ws = (char*)d_ws;
  const size_t ACT = (size_t)NTOK * Cc * 2;       // 25,690,112 B (bf16 act)
  bf16* wqkvT = (bf16*)(ws + 0);                  //  98,304
  bf16* wpT   = (bf16*)(ws + 98304);              //  32,768
  bf16* w1T   = (bf16*)(ws + 131072);             // 131,072
  bf16* w2T   = (bf16*)(ws + 262144);             // 131,072
  bf16* xw    = (bf16*)(ws + 393216);             // ACT   (reused as h2)
  bf16* qb    = (bf16*)(ws + 393216 + ACT);       // ACT
  bf16* kb    = (bf16*)(ws + 393216 + 2 * ACT);   // ACT
  bf16* vb    = (bf16*)(ws + 393216 + 3 * ACT);   // ACT
  bf16* obuf  = (bf16*)(ws + 393216 + 4 * ACT);   // ACT
  float* x1   = (float*)(ws + 393216 + 5 * ACT);  // 2*ACT (f32)
  bf16* h2    = xw;                               // reuse (xw dead after K2)
  bf16* m1    = qb;                               // reuse q/k/v/o (4*ACT == NTOK*512*2)

  const int MT = NTOK / 16;                       // 6272 M tiles

  k_convert<<<768, 256, 0, stream>>>(w_qkv, w_proj, w1, w2, wqkvT, wpT, w1T, w2T);
  k_ln1<<<NTOK / 8, 256, 0, stream>>>(x, gamma1, beta1, xw);
  k_qkv<<<dim3(MT, 3), 32, 0, stream>>>(xw, wqkvT, b_qkv, qb, kb, vb);
  k_attn<<<NWTOT, 256, 49152, stream>>>(qb, kb, vb, btab, obuf);
  k_proj<<<MT, 32, 0, stream>>>(obuf, wpT, b_proj, x, x1);
  k_ln2<<<NTOK / 8, 256, 0, stream>>>(x1, gamma2, beta2, h2);
  k_mlp1<<<dim3(MT, 4), 32, 0, stream>>>(h2, w1T, b1, m1);
  k_mlp2<<<MT, 32, 0, stream>>>(m1, w2T, b2, x1, out);
}
